// HEMoEv2_20813411517123
// MI455X (gfx1250) — matrix operand order repaired and checked by reading, not compiled, not run on hardware
//
#include <hip/hip_runtime.h>
#include <hip/hip_bf16.h>
#include <math.h>

typedef __attribute__((ext_vector_type(16))) _Float16 v16h;
typedef __attribute__((ext_vector_type(8)))  _Float16 v8h;
typedef __attribute__((ext_vector_type(8)))  float    v8f;
typedef __attribute__((ext_vector_type(4)))  float    v4f;
typedef __attribute__((ext_vector_type(4)))  int      i32x4;

#define Bb 8
#define Ss 256
#define Dd 256
#define Vv 4096
#define Ee 32
#define Nn (Bb*Ss)            // 2048
#define FAST_RATIO 0.7f
#define INV_2SIG2 (1.0f/8.0f) // 1/(2*SIGMA^2), SIGMA=2

#if defined(__AMDGCN__) && __has_builtin(__builtin_amdgcn_global_load_async_to_lds_b128)
#define HAVE_ASYNC_LDS 1
#else
#define HAVE_ASYNC_LDS 0
#endif

// ---------------- K1: embedding gather + sinusoidal PE + row norms ----------
__global__ __launch_bounds__(256)
void k_embed_pe(const int* __restrict__ x, const float* __restrict__ embed,
                float* __restrict__ hf32, _Float16* __restrict__ hf16,
                float* __restrict__ xnorm) {
  int n = blockIdx.x, d = threadIdx.x;
  int s = n & (Ss - 1);
  float v = embed[(size_t)x[n] * Dd + d];
  int j2 = (d >> 1) << 1;
  float ang = (float)s * expf((float)j2 * (-9.210340371976184f / (float)Dd));
  v += (d & 1) ? cosf(ang) : sinf(ang);
  hf32[n * Dd + d] = v;
  hf16[n * Dd + d] = (_Float16)v;
  __shared__ float red[256];
  red[d] = v * v;
  __syncthreads();
  for (int ofs = 128; ofs > 0; ofs >>= 1) {
    if (d < ofs) red[d] += red[d + ofs];
    __syncthreads();
  }
  if (d == 0) xnorm[n] = red[0];
}

// ---------------- K1b: mu -> f16 + norms ------------------------------------
__global__ __launch_bounds__(256)
void k_mu16(const float* __restrict__ mu, _Float16* __restrict__ mu16,
            float* __restrict__ munorm) {
  int e = blockIdx.x, d = threadIdx.x;
  float m = mu[e * Dd + d];
  mu16[e * Dd + d] = (_Float16)m;
  __shared__ float red[256];
  red[d] = m * m;
  __syncthreads();
  for (int ofs = 128; ofs > 0; ofs >>= 1) {
    if (d < ofs) red[d] += red[d + ofs];
    __syncthreads();
  }
  if (d == 0) munorm[e] = red[0];
}

// ---------------- K2: scores via WMMA (16 rows x 32 experts per wave) -------
__global__ __launch_bounds__(32)
void k_scores(const _Float16* __restrict__ hf16, const _Float16* __restrict__ mu16,
              const float* __restrict__ xnorm, const float* __restrict__ munorm,
              const float* __restrict__ charge,
              float* __restrict__ scores, float* __restrict__ colsum) {
  int n0 = blockIdx.x * 16;
  int lane = threadIdx.x;
  int mrow = lane & 15;
  int hi = lane >> 4;
  const int kA0 = hi * 16;
  const int kB0 = hi * 16;
  v8f c0 = {}, c1 = {};
  for (int kk = 0; kk < Dd; kk += 32) {
    union { v16h v; v8h h[2]; } a;
    const _Float16* arow = hf16 + (size_t)(n0 + mrow) * Dd + kk;
    a.h[0] = *reinterpret_cast<const v8h*>(arow + kA0);
    a.h[1] = *reinterpret_cast<const v8h*>(arow + kA0 + 8);
    v16h b0 = *reinterpret_cast<const v16h*>(mu16 + (size_t)mrow * Dd + kk + kB0);
    v16h b1 = *reinterpret_cast<const v16h*>(mu16 + (size_t)(16 + mrow) * Dd + kk + kB0);
    c0 = __builtin_amdgcn_wmma_f32_16x16x32_f16(false, a.v, false, b0, (short)0, c0, false, false);
    c1 = __builtin_amdgcn_wmma_f32_16x16x32_f16(false, a.v, false, b1, (short)0, c1, false, false);
  }
  int e0 = mrow, e1 = 16 + mrow;
  float mn0 = munorm[e0], mn1 = munorm[e1];
  float ch0 = charge[e0], ch1 = charge[e1];
  float p0 = 0.f, p1 = 0.f;
#pragma unroll
  for (int v = 0; v < 8; ++v) {
    int row = n0 + v + hi * 8;
    float xn = xnorm[row];
    float s0 = fmaxf(xn + mn0 - 2.f * c0[v], 0.f);
    float s1 = fmaxf(xn + mn1 - 2.f * c1[v], 0.f);
    float sc0 = expf(-s0 * INV_2SIG2) * ch0;
    float sc1 = expf(-s1 * INV_2SIG2) * ch1;
    scores[row * Ee + e0] = sc0;
    scores[row * Ee + e1] = sc1;
    p0 += sc0; p1 += sc1;
  }
  p0 += __shfl_xor(p0, 16, 32);
  p1 += __shfl_xor(p1, 16, 32);
  if (hi == 0) {
    atomicAdd(&colsum[e0], p0);
    atomicAdd(&colsum[e1], p1);
  }
}

// ---------------- K3: top-2 over expert mean scores -------------------------
__global__ void k_topk(const float* __restrict__ colsum, int* __restrict__ topidx) {
  if (threadIdx.x == 0) {
    int b0 = 0; float v0 = colsum[0];
    for (int e = 1; e < Ee; ++e) { float v = colsum[e]; if (v > v0) { v0 = v; b0 = e; } }
    int b1 = (b0 == 0) ? 1 : 0; float v1 = colsum[b1];
    for (int e = 0; e < Ee; ++e) {
      if (e == b0 || e == b1) continue;
      float v = colsum[e]; if (v > v1) { v1 = v; b1 = e; }
    }
    topidx[0] = b0; topidx[1] = b1;
  }
}

// ---------------- K4: gather selected expert weights -> f16 -----------------
__global__ __launch_bounds__(256)
void k_wk16(const float* __restrict__ expert_w, const int* __restrict__ topidx,
            _Float16* __restrict__ wk16) {
  int k = blockIdx.y;
  int idx = topidx[k];
  int t = blockIdx.x * 256 + threadIdx.x;
  wk16[(size_t)k * Dd * Dd + t] = (_Float16)expert_w[(size_t)idx * Dd * Dd + t];
}

// ---------------- K5: eo[n,k,o] = sum_i hf[n,i]*Wk[k,o,i] via WMMA ----------
__global__ __launch_bounds__(128)
void k_eo(const _Float16* __restrict__ hf16, const _Float16* __restrict__ wk16,
          float* __restrict__ eo, float* __restrict__ eocs) {
  int k = blockIdx.z;
  int wave = threadIdx.x >> 5;
  int lane = threadIdx.x & 31;
  int n0 = blockIdx.x * 64 + wave * 16;
  int c0 = blockIdx.y * 64;
  int mrow = lane & 15, hi = lane >> 4;
  const int kA0 = hi * 16, kB0 = hi * 16;
  const _Float16* wbase = wk16 + (size_t)k * Dd * Dd;
  v8f acc[4] = {{}, {}, {}, {}};
  for (int kk = 0; kk < Dd; kk += 32) {
    union { v16h v; v8h h[2]; } a;
    const _Float16* arow = hf16 + (size_t)(n0 + mrow) * Dd + kk;
    __builtin_prefetch(arow + 32, 0, 0);
    a.h[0] = *reinterpret_cast<const v8h*>(arow + kA0);
    a.h[1] = *reinterpret_cast<const v8h*>(arow + kA0 + 8);
#pragma unroll
    for (int j = 0; j < 4; ++j) {
      const _Float16* brow = wbase + (size_t)(c0 + j * 16 + mrow) * Dd + kk + kB0;
      __builtin_prefetch(brow + 32, 0, 0);
      v16h b = *reinterpret_cast<const v16h*>(brow);
      acc[j] = __builtin_amdgcn_wmma_f32_16x16x32_f16(false, a.v, false, b, (short)0, acc[j], false, false);
    }
  }
#pragma unroll
  for (int j = 0; j < 4; ++j) {
    int col = c0 + j * 16 + mrow;
    float part = 0.f;
#pragma unroll
    for (int v = 0; v < 8; ++v) {
      int row = n0 + v + hi * 8;
      float val = acc[j][v];
      eo[(size_t)row * (2 * Dd) + k * Dd + col] = val;
      part += val;
    }
    part += __shfl_xor(part, 16, 32);
    if (hi == 0) atomicAdd(&eocs[k * Dd + col], part);
  }
}

// ---------------- K6: h_new = gate*expert_h + (1-gate)*mean(eo) -------------
__global__ void k_hnew(const float* __restrict__ expert_gate, const float* __restrict__ expert_h,
                       const int* __restrict__ topidx, const float* __restrict__ eocs,
                       float* __restrict__ hnew) {
  int t = blockIdx.x * blockDim.x + threadIdx.x;
  if (t >= 2 * Dd) return;
  int k = t >> 8, d = t & 255;
  int idx = topidx[k];
  float g = 1.f / (1.f + expf(-expert_gate[idx]));
  float mean = eocs[k * Dd + d] * (1.f / (float)Nn);
  hnew[t] = g * expert_h[(size_t)idx * Dd + d] + (1.f - g) * mean;
}

// ============ L1-distance tile machinery (async double-buffered) ============
// Tile: 32 rows x 64 cols per 256-thread block; K staged in 64-wide chunks.
// Row-major LDS tiles (pitch 68) fetched with per-lane
// GLOBAL_LOAD_ASYNC_TO_LDS_B128 (6 per thread per chunk), double-buffered with
// s_wait_asynccnt; compute vectorizes along i (6x ds_load_b128 per 32 ops).

__device__ __forceinline__ void wait_async_prev(int more) {
#if HAVE_ASYNC_LDS
#if __has_builtin(__builtin_amdgcn_s_wait_asynccnt)
  if (more) __builtin_amdgcn_s_wait_asynccnt(6);
  else      __builtin_amdgcn_s_wait_asynccnt(0);
#else
  if (more) asm volatile("s_wait_asynccnt 0x6" ::: "memory");
  else      asm volatile("s_wait_asynccnt 0x0" ::: "memory");
#endif
#else
  (void)more;
#endif
}

__device__ __forceinline__ void issue_tile(const float* __restrict__ X,
                                           const float* __restrict__ W,
                                           int n0, int cb, int ic,
                                           float (&Xs)[32][68], float (&Ws)[64][68],
                                           int tid) {
#pragma unroll
  for (int t = 0; t < 6; ++t) {
    int idx = t * 256 + tid;               // 0..1535 : 1024 W-transfers + 512 X
    const float* gp;
    float* lp;
    if (idx < 1024) {
      int c = idx >> 4, u = idx & 15;
      gp = W + (size_t)(cb + c) * Dd + ic + u * 4;
      lp = &Ws[c][u * 4];
    } else {
      int r = (idx - 1024) >> 4, u = idx & 15;
      gp = X + (size_t)(n0 + r) * Dd + ic + u * 4;
      lp = &Xs[r][u * 4];
    }
#if HAVE_ASYNC_LDS
    __builtin_amdgcn_global_load_async_to_lds_b128(
        (__attribute__((address_space(1))) i32x4*)gp,
        (__attribute__((address_space(3))) i32x4*)lp,
        0, 0);
#else
    *reinterpret_cast<v4f*>(lp) = *reinterpret_cast<const v4f*>(gp);
#endif
  }
}

__device__ __forceinline__ void l1_compute(const float (&Xs)[32][68], const float (&Ws)[64][68],
                                           int rg, int cg, float acc[2][4]) {
#pragma unroll 2
  for (int i = 0; i < 64; i += 4) {
    v4f xv0 = *reinterpret_cast<const v4f*>(&Xs[rg * 2 + 0][i]);
    v4f xv1 = *reinterpret_cast<const v4f*>(&Xs[rg * 2 + 1][i]);
    v4f wv0 = *reinterpret_cast<const v4f*>(&Ws[cg * 4 + 0][i]);
    v4f wv1 = *reinterpret_cast<const v4f*>(&Ws[cg * 4 + 1][i]);
    v4f wv2 = *reinterpret_cast<const v4f*>(&Ws[cg * 4 + 2][i]);
    v4f wv3 = *reinterpret_cast<const v4f*>(&Ws[cg * 4 + 3][i]);
#pragma unroll
    for (int ii = 0; ii < 4; ++ii) {
      float x0 = xv0[ii], x1 = xv1[ii];
      acc[0][0] += fabsf(x0 - wv0[ii]); acc[0][1] += fabsf(x0 - wv1[ii]);
      acc[0][2] += fabsf(x0 - wv2[ii]); acc[0][3] += fabsf(x0 - wv3[ii]);
      acc[1][0] += fabsf(x1 - wv0[ii]); acc[1][1] += fabsf(x1 - wv1[ii]);
      acc[1][2] += fabsf(x1 - wv2[ii]); acc[1][3] += fabsf(x1 - wv3[ii]);
    }
  }
}

// ---------------- K7: h_merged = clip(0.5+0.25*(-L1(hf,w1)+b1)) + 0.3*slow --
__global__ __launch_bounds__(256)
void k_merge(const float* __restrict__ hf32, const float* __restrict__ w1,
             const float* __restrict__ b1, const float* __restrict__ scores,
             const int* __restrict__ topidx, const float* __restrict__ eo,
             const float* __restrict__ hnew, float* __restrict__ hmerged) {
  __shared__ __align__(16) float Xs[2][32][68];
  __shared__ __align__(16) float Ws[2][64][68];
  int n0 = blockIdx.x * 32;
  int cb = blockIdx.y * 64;
  int tid = threadIdx.x;
  int rg = tid >> 4, cg = tid & 15;
  float acc[2][4] = {};
  issue_tile(hf32, w1, n0, cb, 0, Xs[0], Ws[0], tid);
  for (int ch = 0; ch < Dd / 64; ++ch) {
    if (ch < Dd / 64 - 1)
      issue_tile(hf32, w1, n0, cb, (ch + 1) * 64, Xs[(ch + 1) & 1], Ws[(ch + 1) & 1], tid);
    wait_async_prev(ch < Dd / 64 - 1);
    __syncthreads();
    l1_compute(Xs[ch & 1], Ws[ch & 1], rg, cg, acc);
    __syncthreads();
  }
  int ti0 = topidx[0], ti1 = topidx[1];
#pragma unroll
  for (int rr = 0; rr < 2; ++rr) {
    int n = n0 + rg * 2 + rr;
    float w0 = scores[n * Ee + ti0];
    float wn1 = scores[n * Ee + ti1];
#pragma unroll
    for (int cc = 0; cc < 4; ++cc) {
      int j = cb + cg * 4 + cc;
      float hfast = 0.5f + 0.25f * (-acc[rr][cc] + b1[j]);
      hfast = fminf(fmaxf(hfast, 0.f), 1.f);
      float slow = w0  * (eo[(size_t)n * (2 * Dd) + j]      + hnew[j]) +
                   wn1 * (eo[(size_t)n * (2 * Dd) + Dd + j] + hnew[Dd + j]);
      hmerged[(size_t)n * Dd + j] = hfast + (1.f - FAST_RATIO) * slow;
    }
  }
}

// ---------------- K8: logits = -L1(h_merged, w2) + b2 -----------------------
__global__ __launch_bounds__(256)
void k_logits(const float* __restrict__ hm, const float* __restrict__ w2,
              const float* __restrict__ b2, float* __restrict__ out) {
  __shared__ __align__(16) float Xs[2][32][68];
  __shared__ __align__(16) float Ws[2][64][68];
  int n0 = blockIdx.x * 32;
  int cb = blockIdx.y * 64;
  int tid = threadIdx.x;
  int rg = tid >> 4, cg = tid & 15;
  float acc[2][4] = {};
  issue_tile(hm, w2, n0, cb, 0, Xs[0], Ws[0], tid);
  for (int ch = 0; ch < Dd / 64; ++ch) {
    if (ch < Dd / 64 - 1)
      issue_tile(hm, w2, n0, cb, (ch + 1) * 64, Xs[(ch + 1) & 1], Ws[(ch + 1) & 1], tid);
    wait_async_prev(ch < Dd / 64 - 1);
    __syncthreads();
    l1_compute(Xs[ch & 1], Ws[ch & 1], rg, cg, acc);
    __syncthreads();
  }
#pragma unroll
  for (int rr = 0; rr < 2; ++rr) {
    int n = n0 + rg * 2 + rr;
#pragma unroll
    for (int cc = 0; cc < 4; ++cc) {
      int v = cb + cg * 4 + cc;
      out[(size_t)n * Vv + v] = -acc[rr][cc] + b2[v];
    }
  }
}

// ---------------------------------------------------------------------------
extern "C" void kernel_launch(void* const* d_in, const int* in_sizes, int n_in,
                              void* d_out, int out_size, void* d_ws, size_t ws_size,
                              hipStream_t stream) {
  const int*   x        = (const int*)  d_in[0];
  const float* embed    = (const float*)d_in[1];
  const float* w1       = (const float*)d_in[2];
  const float* b1       = (const float*)d_in[3];
  const float* w2       = (const float*)d_in[4];
  const float* b2       = (const float*)d_in[5];
  const float* mu       = (const float*)d_in[6];
  const float* expert_w = (const float*)d_in[7];
  const float* e_gate   = (const float*)d_in[8];
  const float* e_charge = (const float*)d_in[9];
  const float* e_h      = (const float*)d_in[10];
  float* out = (float*)d_out;

  char* ws = (char*)d_ws;
  size_t off = 0;
  auto up = [&](size_t bytes) { size_t o = off; off = (off + bytes + 511) & ~(size_t)511; return o; };
  size_t o_hf32  = up((size_t)Nn * Dd * 4);
  size_t o_hf16  = up((size_t)Nn * Dd * 2);
  size_t o_mu16  = up((size_t)Ee * Dd * 2);
  size_t o_xn    = up((size_t)Nn * 4);
  size_t o_mn    = up((size_t)Ee * 4);
  size_t o_sc    = up((size_t)Nn * Ee * 4);
  size_t o_cs    = up((size_t)Ee * 4);
  size_t o_ti    = up(2 * 4);
  size_t o_wk16  = up((size_t)2 * Dd * Dd * 2);
  size_t o_eo    = up((size_t)Nn * 2 * Dd * 4);
  size_t o_eocs  = up((size_t)2 * Dd * 4);
  size_t o_hnew  = up((size_t)2 * Dd * 4);
  size_t o_hm    = up((size_t)Nn * Dd * 4);
  (void)ws_size; (void)n_in; (void)in_sizes; (void)out_size;

  float*    hf32   = (float*)   (ws + o_hf32);
  _Float16* hf16   = (_Float16*)(ws + o_hf16);
  _Float16* mu16   = (_Float16*)(ws + o_mu16);
  float*    xnorm  = (float*)   (ws + o_xn);
  float*    munorm = (float*)   (ws + o_mn);
  float*    scores = (float*)   (ws + o_sc);
  float*    colsum = (float*)   (ws + o_cs);
  int*      topidx = (int*)     (ws + o_ti);
  _Float16* wk16   = (_Float16*)(ws + o_wk16);
  float*    eo     = (float*)   (ws + o_eo);
  float*    eocs   = (float*)   (ws + o_eocs);
  float*    hnew   = (float*)   (ws + o_hnew);
  float*    hm     = (float*)   (ws + o_hm);

  (void)hipMemsetAsync(colsum, 0, Ee * 4, stream);
  (void)hipMemsetAsync(eocs, 0, 2 * Dd * 4, stream);

  k_embed_pe<<<Nn, 256, 0, stream>>>(x, embed, hf32, hf16, xnorm);
  k_mu16<<<Ee, 256, 0, stream>>>(mu, mu16, munorm);
  k_scores<<<Nn / 16, 32, 0, stream>>>(hf16, mu16, xnorm, munorm, e_charge, scores, colsum);
  k_topk<<<1, 32, 0, stream>>>(colsum, topidx);
  k_wk16<<<dim3(Dd * Dd / 256, 2), 256, 0, stream>>>(expert_w, topidx, wk16);
  k_eo<<<dim3(Nn / 64, Dd / 64, 2), 128, 0, stream>>>(hf16, wk16, eo, eocs);
  k_hnew<<<2, 256, 0, stream>>>(e_gate, e_h, topidx, eocs, hnew);
  k_merge<<<dim3(Nn / 32, Dd / 64), 256, 0, stream>>>(hf32, w1, b1, scores, topidx, eo, hnew, hm);
  k_logits<<<dim3(Nn / 32, Vv / 64), 256, 0, stream>>>(hm, w2, b2, out);
}
